// ClassicTransformer_13288628813910
// MI455X (gfx1250) — matrix operand order repaired and checked by reading, not compiled, not run on hardware
//
#include <hip/hip_runtime.h>
#include <hip/hip_bf16.h>

// ---------------- problem constants ----------------
#define VOCAB 50000
#define EMBED 256
#define MODULES 2
#define HEADS 4
#define HID 64
#define BATCH 8
#define SEQ 2048
#define LABELS 16
#define BS (BATCH * SEQ)            // 16384 tokens
#define QKVCOLS (3 * HEADS * HID)   // 768

typedef __attribute__((ext_vector_type(8)))  float  v8f;
typedef __attribute__((ext_vector_type(16))) __bf16 v16bf;

union FragU { unsigned u[8]; v16bf v; };

// ---------------- helpers ----------------
static __device__ inline unsigned short f2bf(float x) {
  unsigned u = __builtin_bit_cast(unsigned, x);
  u = (u + 0x7fffu + ((u >> 16) & 1u)) >> 16;
  return (unsigned short)u;
}

// A fragment: 16x32 bf16, tile row-major [16][32] u16
static __device__ inline void load_a_frag(const unsigned short* t, int lane, unsigned u[8]) {
  const int m  = lane & 15;
  const int kh = (lane >> 4) << 3;
#pragma unroll
  for (int i = 0; i < 8; ++i) {
    const int k = (kh << 1) + (i << 1);
    u[i] = *(const unsigned*)(t + m * 32 + k);      // packs (k, k+1)
  }
}

// B fragment: 32x16 bf16 from LDS tile stored [N][32] row-major
static __device__ inline void load_b_frag_nk(const unsigned short* t, int lane, unsigned u[8]) {
  const int n  = lane & 15;
  const int kb = (lane >> 4) << 4;
#pragma unroll
  for (int i = 0; i < 8; ++i) {
    const int k = kb + (i << 1);
    u[i] = *(const unsigned*)(t + n * 32 + k);
  }
}

// B fragment: 32x16 bf16 from LDS tile stored [K=32][64] row-major
static __device__ inline void load_b_frag_kn(const unsigned short* t, int lane, unsigned u[8]) {
  const int n  = lane & 15;
  const int kb = (lane >> 4) << 4;
#pragma unroll
  for (int i = 0; i < 8; ++i) {
    const int k = kb + (i << 1);
    unsigned lo = t[(k)     * 64 + n];
    unsigned hi = t[(k + 1) * 64 + n];
    u[i] = lo | (hi << 16);
  }
}

// Block computes a 64x64 f32 tile of A(MxK) x B(KxN); 8 waves in a 4x2 grid,
// each wave owns a 16x32 tile (two accumulators sharing one A fragment).
// Double-buffered LDS + register prefetch: next chunk's global loads are in
// flight while the current chunk's WMMAs execute; one barrier per chunk.
// A row-major [.,lda]; B is [N][K] row-major (B_IS_NK) or [K][N] row-major.
// Pointers pre-offset to the block's row/col origin. K % 32 == 0.
template <bool B_IS_NK>
static __device__ inline void gemm64x64(const unsigned short* __restrict__ A, int lda,
                                        const unsigned short* __restrict__ Bsrc, int ldb,
                                        int K, unsigned short* ldsA, unsigned short* ldsB,
                                        v8f& acc0, v8f& acc1, int& m0, int& n0) {
  const int tid  = threadIdx.x;
  const int lane = tid & 31;
  const int wave = tid >> 5;
  m0 = (wave & 3) << 4;        // 4 row tiles of 16
  n0 = (wave >> 2) << 5;       // 2 col tiles of 32
  acc0 = (v8f){0.f, 0.f, 0.f, 0.f, 0.f, 0.f, 0.f, 0.f};
  acc1 = acc0;

  // staging coordinates (each of 256 threads moves one uint4 per tile)
  const int ar = tid >> 2, ak = (tid & 3) << 3;                    // A: 64 x 32
  const int br = B_IS_NK ? (tid >> 2) : (tid >> 3);                // B: 64n x 32k or 32k x 64n
  const int bc = B_IS_NK ? ((tid & 3) << 3) : ((tid & 7) << 3);

  // prologue prefetch of chunk 0
  uint4 ra = *(const uint4*)(A + (size_t)ar * lda + ak);
  uint4 rb = B_IS_NK ? *(const uint4*)(Bsrc + (size_t)br * ldb + bc)
                     : *(const uint4*)(Bsrc + (size_t)br * ldb + bc);

  for (int kb = 0, it = 0; kb < K; kb += 32, ++it) {
    unsigned short* sA = ldsA + (it & 1) * (64 * 32);
    unsigned short* sB = ldsB + (it & 1) * (64 * 32);
    *(uint4*)(sA + ar * 32 + ak) = ra;
    if (B_IS_NK) *(uint4*)(sB + br * 32 + bc) = rb;
    else         *(uint4*)(sB + br * 64 + bc) = rb;

    // prefetch chunk kb+32 while this chunk's WMMAs run
    const int kn = kb + 32;
    if (kn < K) {
      ra = *(const uint4*)(A + (size_t)ar * lda + kn + ak);
      if (B_IS_NK) rb = *(const uint4*)(Bsrc + (size_t)br * ldb + kn + bc);
      else         rb = *(const uint4*)(Bsrc + (size_t)(kn + br) * ldb + bc);
    }
    __syncthreads();

    FragU a, b0, b1;
    load_a_frag(sA + m0 * 32, lane, a.u);
    if (B_IS_NK) {
      load_b_frag_nk(sB + (n0)      * 32, lane, b0.u);
      load_b_frag_nk(sB + (n0 + 16) * 32, lane, b1.u);
    } else {
      load_b_frag_kn(sB + n0,      lane, b0.u);
      load_b_frag_kn(sB + n0 + 16, lane, b1.u);
    }
    acc0 = __builtin_amdgcn_wmma_f32_16x16x32_bf16(false, a.v, false, b0.v,
                                                   (short)0, acc0, false, false);
    acc1 = __builtin_amdgcn_wmma_f32_16x16x32_bf16(false, a.v, false, b1.v,
                                                   (short)0, acc1, false, false);
    // no trailing barrier: next iteration writes the other LDS stage; the
    // s_wait_dscnt before these WMMAs orders reads ahead of the next barrier.
  }
}

// ---------------- kernels ----------------

__global__ void __launch_bounds__(64) ct_gather_kernel(const int* __restrict__ seqs,
                                                       const float* __restrict__ emb,
                                                       float* __restrict__ x) {
  const int tok = blockIdx.x;
  const int v = seqs[tok];
  const float4* src = (const float4*)(emb + (size_t)v * EMBED);
  float4* dst = (float4*)(x + (size_t)tok * EMBED);
  dst[threadIdx.x] = src[threadIdx.x];
}

__global__ void __launch_bounds__(256) ct_pack_qkv_kernel(const float* __restrict__ Wq,
                                                          const float* __restrict__ Wk,
                                                          const float* __restrict__ Wv,
                                                          const float* __restrict__ bq,
                                                          const float* __restrict__ bk,
                                                          const float* __restrict__ bv,
                                                          int mod, unsigned short* __restrict__ Wout,
                                                          float* __restrict__ bout) {
  const int idx = blockIdx.x * 256 + threadIdx.x;           // 768*256 elems
  const int r = idx >> 8, e = idx & 255;
  const int mat = r >> 8, rr = r & 255;
  const float* W = (mat == 0) ? Wq : (mat == 1) ? Wk : Wv;
  Wout[idx] = f2bf(W[(size_t)mod * (HEADS * HID * EMBED) + (size_t)rr * EMBED + e]);
  if (idx < QKVCOLS) {
    const int m2 = idx >> 8, r2 = idx & 255;
    const float* bb = (m2 == 0) ? bq : (m2 == 1) ? bk : bv;
    bout[idx] = bb[mod * (HEADS * HID) + r2];
  }
}

__global__ void __launch_bounds__(256) ct_pack_wz_kernel(const float* __restrict__ Wz, int mod,
                                                         unsigned short* __restrict__ out) {
  const int idx = blockIdx.x * 256 + threadIdx.x;           // 256*256 elems
  out[idx] = f2bf(Wz[(size_t)mod * EMBED * (HEADS * HID) + idx]);
}

__global__ void __launch_bounds__(256) ct_tobf_kernel(const float* __restrict__ x,
                                                      unsigned short* __restrict__ xbf) {
  const int idx = blockIdx.x * 256 + threadIdx.x;
  xbf[idx] = f2bf(x[idx]);
}

// QKV = relu(x @ Wqkv^T + b); scatter to Q/K/V [B,H,S,HID] bf16
__global__ void __launch_bounds__(256) ct_qkv_gemm_kernel(const unsigned short* __restrict__ xbf,
                                                          const unsigned short* __restrict__ W,
                                                          const float* __restrict__ bias,
                                                          unsigned short* __restrict__ Qb,
                                                          unsigned short* __restrict__ Kb,
                                                          unsigned short* __restrict__ Vb) {
  __shared__ __align__(16) unsigned short ldsA[2 * 64 * 32];
  __shared__ __align__(16) unsigned short ldsB[2 * 64 * 32];
  const unsigned short* A  = xbf + (size_t)blockIdx.x * 64 * EMBED;
  const unsigned short* Bp = W   + (size_t)blockIdx.y * 64 * EMBED;
  int m0, n0;
  v8f acc[2];
  gemm64x64<true>(A, EMBED, Bp, EMBED, EMBED, ldsA, ldsB, acc[0], acc[1], m0, n0);
  const int lane = threadIdx.x & 31;
  const int row0 = blockIdx.x * 64 + m0;
#pragma unroll
  for (int half = 0; half < 2; ++half) {
    const int col0 = blockIdx.y * 64 + n0 + half * 16;
#pragma unroll
    for (int i = 0; i < 8; ++i) {
      const int mrow = i + ((lane >> 4) << 3);
      const int n = lane & 15;
      const int tok = row0 + mrow, col = col0 + n;
      float v = acc[half][i] + bias[col];
      v = v > 0.f ? v : 0.f;
      const int mat = col >> 8, h = (col >> 6) & 3, d = col & 63;
      unsigned short* dst = (mat == 0) ? Qb : (mat == 1) ? Kb : Vb;
      const int b = tok >> 11, s = tok & (SEQ - 1);
      dst[(((size_t)b * HEADS + h) * SEQ + s) * HID + d] = f2bf(v);
    }
  }
}

// Sc[h][s][t] = (Q[h,s,:] . K[h,t,:]) / 16
__global__ void __launch_bounds__(256) ct_qk_gemm_kernel(const unsigned short* __restrict__ Q,
                                                         const unsigned short* __restrict__ Km,
                                                         float* __restrict__ Sc) {
  __shared__ __align__(16) unsigned short ldsA[2 * 64 * 32];
  __shared__ __align__(16) unsigned short ldsB[2 * 64 * 32];
  const int h = blockIdx.z;
  const unsigned short* A  = Q  + ((size_t)h * SEQ + (size_t)blockIdx.x * 64) * HID;
  const unsigned short* Bp = Km + ((size_t)h * SEQ + (size_t)blockIdx.y * 64) * HID;
  int m0, n0;
  v8f acc[2];
  gemm64x64<true>(A, HID, Bp, HID, HID, ldsA, ldsB, acc[0], acc[1], m0, n0);
  const int lane = threadIdx.x & 31;
  const int s0 = blockIdx.x * 64 + m0;
#pragma unroll
  for (int half = 0; half < 2; ++half) {
    const int t0 = blockIdx.y * 64 + n0 + half * 16;
#pragma unroll
    for (int i = 0; i < 8; ++i) {
      const int mrow = i + ((lane >> 4) << 3);
      const int n = lane & 15;
      Sc[((size_t)h * SEQ + s0 + mrow) * SEQ + t0 + n] = acc[half][i] * 0.0625f;
    }
  }
}

// softmax over the query axis (rows s of Sc[h][:,t]) — online max/sum, 2 passes
__global__ void __launch_bounds__(256) ct_softmax_col_kernel(float* __restrict__ Sc) {
  const int t = blockIdx.x * 256 + threadIdx.x;
  float* base = Sc + (size_t)blockIdx.y * SEQ * SEQ;
  float mx = -3.402823e38f, sum = 0.f;
  for (int s = 0; s < SEQ; ++s) {
    const float v = base[(size_t)s * SEQ + t];
    const float nm = fmaxf(mx, v);
    sum = sum * __expf(mx - nm) + __expf(v - nm);
    mx = nm;
  }
  const float inv = 1.f / sum;
  for (int s = 0; s < SEQ; ++s) {
    const size_t idx = (size_t)s * SEQ + t;
    base[idx] = __expf(base[idx] - mx) * inv;
  }
}

// softmax over the key axis (row-contiguous), emits bf16 probs for A*V
__global__ void __launch_bounds__(256) ct_softmax_row_kernel(const float* __restrict__ Sc,
                                                             unsigned short* __restrict__ Pb) {
  __shared__ float red[256];
  const int h = blockIdx.y, srow = blockIdx.x, tid = threadIdx.x;
  const float* row = Sc + ((size_t)h * SEQ + srow) * SEQ;
  float mx = -3.402823e38f;
  for (int t = tid; t < SEQ; t += 256) mx = fmaxf(mx, row[t]);
  red[tid] = mx; __syncthreads();
  for (int o = 128; o > 0; o >>= 1) { if (tid < o) red[tid] = fmaxf(red[tid], red[tid + o]); __syncthreads(); }
  mx = red[0]; __syncthreads();
  float sum = 0.f;
  for (int t = tid; t < SEQ; t += 256) sum += __expf(row[t] - mx);
  red[tid] = sum; __syncthreads();
  for (int o = 128; o > 0; o >>= 1) { if (tid < o) red[tid] += red[tid + o]; __syncthreads(); }
  const float inv = 1.f / red[0];
  unsigned short* prow = Pb + ((size_t)h * SEQ + srow) * SEQ;
  for (int t = tid; t < SEQ; t += 256) prow[t] = f2bf(__expf(row[t] - mx) * inv);
}

// Z[h][s][d] = sum_t P[h][s][t] * V[h][t][d]; write Zall[s][h*64+d]
__global__ void __launch_bounds__(256) ct_av_gemm_kernel(const unsigned short* __restrict__ Pb,
                                                         const unsigned short* __restrict__ V,
                                                         unsigned short* __restrict__ Zall) {
  __shared__ __align__(16) unsigned short ldsA[2 * 64 * 32];
  __shared__ __align__(16) unsigned short ldsB[2 * 64 * 32];
  const int h = blockIdx.z;
  const unsigned short* A  = Pb + ((size_t)h * SEQ + (size_t)blockIdx.x * 64) * SEQ;
  const unsigned short* Bp = V  + (size_t)h * SEQ * HID;   // full 64-col N block
  int m0, n0;
  v8f acc[2];
  gemm64x64<false>(A, SEQ, Bp, HID, SEQ, ldsA, ldsB, acc[0], acc[1], m0, n0);
  const int lane = threadIdx.x & 31;
  const int s0 = blockIdx.x * 64 + m0;
#pragma unroll
  for (int half = 0; half < 2; ++half) {
    const int d0 = n0 + half * 16;
#pragma unroll
    for (int i = 0; i < 8; ++i) {
      const int mrow = i + ((lane >> 4) << 3);
      const int n = lane & 15;
      Zall[(size_t)(s0 + mrow) * (HEADS * HID) + h * HID + (d0 + n)] = f2bf(acc[half][i]);
    }
  }
}

// x += Zall @ Wz^T + bz  (residual)
__global__ void __launch_bounds__(256) ct_zf_gemm_kernel(const unsigned short* __restrict__ Zall,
                                                         const unsigned short* __restrict__ Wzb,
                                                         const float* __restrict__ bz,
                                                         float* __restrict__ x) {
  __shared__ __align__(16) unsigned short ldsA[2 * 64 * 32];
  __shared__ __align__(16) unsigned short ldsB[2 * 64 * 32];
  const unsigned short* A  = Zall + (size_t)blockIdx.x * 64 * (HEADS * HID);
  const unsigned short* Bp = Wzb  + (size_t)blockIdx.y * 64 * (HEADS * HID);
  int m0, n0;
  v8f acc[2];
  gemm64x64<true>(A, HEADS * HID, Bp, HEADS * HID, HEADS * HID, ldsA, ldsB, acc[0], acc[1], m0, n0);
  const int lane = threadIdx.x & 31;
  const int row0 = blockIdx.x * 64 + m0;
#pragma unroll
  for (int half = 0; half < 2; ++half) {
    const int col0 = blockIdx.y * 64 + n0 + half * 16;
#pragma unroll
    for (int i = 0; i < 8; ++i) {
      const int mrow = i + ((lane >> 4) << 3);
      const int n = lane & 15;
      const int tok = row0 + mrow, e = col0 + n;
      x[(size_t)tok * EMBED + e] += acc[half][i] + bz[e];
    }
  }
}

// y[b][l] = sigmoid(x[b].flat . Wo[l] + bo[l])
__global__ void __launch_bounds__(256) ct_final_kernel(const float* __restrict__ x,
                                                       const float* __restrict__ Wo,
                                                       const float* __restrict__ bo,
                                                       float* __restrict__ out) {
  __shared__ float red[256];
  const int b = blockIdx.x >> 4, l = blockIdx.x & 15;
  const float* xr = x  + (size_t)b * (SEQ * EMBED);
  const float* wr = Wo + (size_t)l * (SEQ * EMBED);
  float acc = 0.f;
  for (int i = threadIdx.x; i < SEQ * EMBED; i += 256) acc += xr[i] * wr[i];
  red[threadIdx.x] = acc; __syncthreads();
  for (int o = 128; o > 0; o >>= 1) { if (threadIdx.x < o) red[threadIdx.x] += red[threadIdx.x + o]; __syncthreads(); }
  if (threadIdx.x == 0) {
    const float v = red[0] + bo[l];
    out[b * LABELS + l] = 1.f / (1.f + __expf(-v));
  }
}

// ---------------- host orchestration ----------------
extern "C" void kernel_launch(void* const* d_in, const int* in_sizes, int n_in,
                              void* d_out, int out_size, void* d_ws, size_t ws_size,
                              hipStream_t stream) {
  (void)in_sizes; (void)n_in; (void)out_size; (void)ws_size;
  const int*   seqs = (const int*)  d_in[0];
  const float* emb  = (const float*)d_in[1];
  const float* Wq   = (const float*)d_in[2];
  const float* bq   = (const float*)d_in[3];
  const float* Wk   = (const float*)d_in[4];
  const float* bk   = (const float*)d_in[5];
  const float* Wv   = (const float*)d_in[6];
  const float* bv   = (const float*)d_in[7];
  const float* Wz   = (const float*)d_in[8];
  const float* bz   = (const float*)d_in[9];
  const float* Wo   = (const float*)d_in[10];
  const float* bo   = (const float*)d_in[11];
  float* y = (float*)d_out;

  // workspace carve (256B aligned)
  char* p = (char*)d_ws;
  auto carve = [&p](size_t bytes) -> void* {
    void* r = (void*)p;
    p += (bytes + 255) & ~(size_t)255;
    return r;
  };
  float*          x    = (float*)         carve((size_t)BS * EMBED * 4);
  unsigned short* xbf  = (unsigned short*)carve((size_t)BS * EMBED * 2);
  unsigned short* Wqkv = (unsigned short*)carve((size_t)QKVCOLS * EMBED * 2);
  float*          bqkv = (float*)         carve((size_t)QKVCOLS * 4);
  unsigned short* Wzb  = (unsigned short*)carve((size_t)EMBED * HEADS * HID * 2);
  unsigned short* Qb   = (unsigned short*)carve((size_t)BATCH * HEADS * SEQ * HID * 2);
  unsigned short* Kb   = (unsigned short*)carve((size_t)BATCH * HEADS * SEQ * HID * 2);
  unsigned short* Vb   = (unsigned short*)carve((size_t)BATCH * HEADS * SEQ * HID * 2);
  float*          Sc   = (float*)         carve((size_t)HEADS * SEQ * SEQ * 4);
  unsigned short* Pb   = (unsigned short*)carve((size_t)HEADS * SEQ * SEQ * 2);
  unsigned short* Zall = (unsigned short*)carve((size_t)BS * HEADS * HID * 2);

  ct_gather_kernel<<<BS, 64, 0, stream>>>(seqs, emb, x);

  for (int mod = 0; mod < MODULES; ++mod) {
    ct_pack_qkv_kernel<<<(QKVCOLS * EMBED) / 256, 256, 0, stream>>>(Wq, Wk, Wv, bq, bk, bv, mod, Wqkv, bqkv);
    ct_pack_wz_kernel<<<(EMBED * HEADS * HID) / 256, 256, 0, stream>>>(Wz, mod, Wzb);
    ct_tobf_kernel<<<(BS * EMBED) / 256, 256, 0, stream>>>(x, xbf);

    ct_qkv_gemm_kernel<<<dim3(BS / 64, QKVCOLS / 64), 256, 0, stream>>>(xbf, Wqkv, bqkv, Qb, Kb, Vb);

    for (int b = 0; b < BATCH; ++b) {
      const unsigned short* Qp = Qb + (size_t)b * HEADS * SEQ * HID;
      const unsigned short* Kp = Kb + (size_t)b * HEADS * SEQ * HID;
      const unsigned short* Vp = Vb + (size_t)b * HEADS * SEQ * HID;
      ct_qk_gemm_kernel<<<dim3(SEQ / 64, SEQ / 64, HEADS), 256, 0, stream>>>(Qp, Kp, Sc);
      ct_softmax_col_kernel<<<dim3(SEQ / 256, HEADS), 256, 0, stream>>>(Sc);
      ct_softmax_row_kernel<<<dim3(SEQ, HEADS), 256, 0, stream>>>(Sc, Pb);
      ct_av_gemm_kernel<<<dim3(SEQ / 64, 1, HEADS), 256, 0, stream>>>(
          Pb, Vp, Zall + (size_t)b * SEQ * HEADS * HID);
    }

    ct_zf_gemm_kernel<<<dim3(BS / 64, EMBED / 64), 256, 0, stream>>>(Zall, Wzb, bz + mod * EMBED, x);
  }

  ct_final_kernel<<<BATCH * LABELS, 256, 0, stream>>>(x, Wo, bo, y);
}
